// NRC_79680233275663
// MI455X (gfx1250) — matrix operand order repaired and checked by reading, not compiled, not run on hardware
//
#include <hip/hip_runtime.h>
#include <hip/hip_bf16.h>

// ---------------------------------------------------------------------------
// NRC / AaD neighborhood loss for MI455X (gfx1250, wave32, WMMA).
// Two fused bf16-WMMA GEMM + streaming-top-6 passes against a 100000x256
// bank (bank converted to bf16 once; 51MB -> resident in 192MB L2).
// ---------------------------------------------------------------------------

typedef __attribute__((ext_vector_type(16))) __bf16 v16bf;
typedef __attribute__((ext_vector_type(8)))  float  v8f;

struct __align__(16) U128 { unsigned int x, y, z, w; };
struct __align__(8)  U64  { unsigned int x, y; };
struct __align__(16) F128 { float x, y, z, w; };

union BFrag { v16bf v; U128 q[2]; };

constexpr int Bq   = 256;      // batch
constexpr int Cc   = 6;        // classes
constexpr int Dd   = 256;      // feature dim
constexpr int Nn   = 100000;   // bank rows
constexpr int KK   = 5;        // K
constexpr int TOPP = 6;        // K+1
constexpr int CHUNK_TILES = 125;   // 125 tiles * 16 keys = 2000 keys per chunk
constexpr int NCHUNKS     = 50;    // 50 * 2000 = 100000
constexpr int QPB         = 128;   // queries per workgroup = 8 waves * 16

__device__ __forceinline__ unsigned short f32_to_bf16(float f) {
  unsigned int u = __float_as_uint(f);
  unsigned int lsb = (u >> 16) & 1u;
  u += 0x7fffu + lsb;                 // round-to-nearest-even
  return (unsigned short)(u >> 16);
}

// Fully branchless sorted insert (desc) into a 6-entry register list.
// Compiles to v_cmp + v_cndmask chain; co-executes with WMMA (TRANS-tracked).
__device__ __forceinline__ void topk_insert(float v, int idx, float* tv, int* ti) {
  float cv = v; int ci = idx;
#pragma unroll
  for (int j = 0; j < TOPP; ++j) {
    bool b = cv > tv[j];
    float nv = b ? cv : tv[j];
    float ov = b ? tv[j] : cv;
    int   ni = b ? ci : ti[j];
    int   oi = b ? ti[j] : ci;
    tv[j] = nv; ti[j] = ni; cv = ov; ci = oi;
  }
}

// --------------------------- small prep kernels ----------------------------

__global__ void nrc_softmax(const float* __restrict__ pred, float* __restrict__ p) {
  int b = blockIdx.x * blockDim.x + threadIdx.x;
  if (b >= Bq) return;
  float x[Cc]; float m = -3.402823e38f;
#pragma unroll
  for (int c = 0; c < Cc; ++c) { x[c] = pred[b * Cc + c]; m = fmaxf(m, x[c]); }
  float s = 0.f;
#pragma unroll
  for (int c = 0; c < Cc; ++c) { x[c] = __expf(x[c] - m); s += x[c]; }
  float inv = 1.f / s;
#pragma unroll
  for (int c = 0; c < Cc; ++c) p[b * Cc + c] = x[c] * inv;
}

__global__ __launch_bounds__(256) void nrc_fnorm(const float* __restrict__ feat,
                                                 unsigned short* __restrict__ outbf) {
  __shared__ float red[256];
  int b = blockIdx.x, t = threadIdx.x;
  float v = feat[b * Dd + t];
  red[t] = v * v;
  __syncthreads();
  for (int s = 128; s > 0; s >>= 1) { if (t < s) red[t] += red[t + s]; __syncthreads(); }
  float denom = fmaxf(sqrtf(red[0]), 1e-12f);
  outbf[b * Dd + t] = f32_to_bf16(v / denom);
}

__global__ void nrc_cvt_bank(const float* __restrict__ src, unsigned short* __restrict__ dst, int n4) {
  int i = blockIdx.x * blockDim.x + threadIdx.x;
  int stride = gridDim.x * blockDim.x;
  for (; i < n4; i += stride) {
    F128 f = reinterpret_cast<const F128*>(src)[i];
    U64 o;
    o.x = (unsigned int)f32_to_bf16(f.x) | ((unsigned int)f32_to_bf16(f.y) << 16);
    o.y = (unsigned int)f32_to_bf16(f.z) | ((unsigned int)f32_to_bf16(f.w) << 16);
    reinterpret_cast<U64*>(dst)[i] = o;
  }
}

__global__ void nrc_copy_f32(const float* __restrict__ src, float* __restrict__ dst, int n) {
  int i = blockIdx.x * blockDim.x + threadIdx.x;
  int stride = gridDim.x * blockDim.x;
  for (; i < n; i += stride) dst[i] = src[i];
}

// fb.at[trg].set(fnorm) / sb.at[trg].set(softmax), deterministic last-write-wins.
__global__ __launch_bounds__(256) void nrc_scatter(const int* __restrict__ trg,
                                                   const unsigned short* __restrict__ fnormbf,
                                                   const float* __restrict__ p,
                                                   unsigned short* __restrict__ fb,
                                                   float* __restrict__ sb) {
  int b = blockIdx.x, t = threadIdx.x;
  int row = trg[b];
  for (int b2 = b + 1; b2 < Bq; ++b2)
    if (trg[b2] == row) return;                       // a later write wins
  fb[(size_t)row * Dd + t] = fnormbf[b * Dd + t];
  if (t < Cc) sb[(size_t)row * Cc + t] = p[b * Cc + t];
}

// ------------------- fused bf16 WMMA GEMM + streaming top-6 ----------------
// D^T tile: M = keys (16/tile, LDS-staged), N = queries (16/wave, registers).
// Each lane owns one query column and 8 key rows of the f32 accumulator ->
// lane-local top-6; lane pairs (L, L+16) merged via LDS at chunk end.

__global__ __launch_bounds__(256) void nrc_gemm_topk(
    const unsigned short* __restrict__ keys,  // fb bf16, [Nn x Dd]
    const unsigned short* __restrict__ qry,   // [Q x Dd] bf16
    float* __restrict__ pvals, int* __restrict__ pidx) {
  __shared__ unsigned short sKey[16 * Dd];    // 8 KB key tile
  __shared__ float sMv[256 * TOPP];
  __shared__ int   sMi[256 * TOPP];

  const int tid   = threadIdx.x;
  const int wave  = tid >> 5;
  const int lane  = tid & 31;
  const int lhalf = lane & 15;
  const bool hiL  = lane >= 16;
  const int chunk = blockIdx.y;
  const int ckey0 = chunk * (CHUNK_TILES * 16);
  const int qg    = blockIdx.x * QPB + wave * 16 + lhalf;   // global query id

  // Preload this lane's B fragments (query data) for all 8 K-steps (K=256).
  // B 32x16 bf16 layout: lane<16 -> K 0-7 & 16-23 of col N=lane, lane>=16 -> K 8-15 & 24-31.
  BFrag bfrag[8];
  {
    const unsigned short* qrow = qry + (size_t)qg * Dd;
#pragma unroll
    for (int kk = 0; kk < 8; ++kk) {
      const U128* ptr = reinterpret_cast<const U128*>(qrow + kk * 32 + (hiL ? 8 : 0));
      bfrag[kk].q[0] = ptr[0];
      bfrag[kk].q[1] = ptr[2];
    }
  }

  float tv[TOPP]; int ti[TOPP];
#pragma unroll
  for (int j = 0; j < TOPP; ++j) { tv[j] = -1e30f; ti[j] = 0x7fffffff; }

  // Register double-buffer for the key tile (2 x 16B per thread).
  U128 stg0, stg1;
  {
    const U128* src = reinterpret_cast<const U128*>(keys + (size_t)ckey0 * Dd);
    stg0 = src[tid];
    stg1 = src[tid + 256];
  }

  for (int t = 0; t < CHUNK_TILES; ++t) {
    const int key0 = ckey0 + t * 16;
    __syncthreads();                                  // prior tile's LDS reads done
    {
      U128* dst = reinterpret_cast<U128*>(sKey);
      dst[tid]       = stg0;
      dst[tid + 256] = stg1;
    }
    if (t + 1 < CHUNK_TILES) {                        // overlap next tile's loads
      const U128* src = reinterpret_cast<const U128*>(keys + (size_t)(key0 + 16) * Dd);
      stg0 = src[tid];
      stg1 = src[tid + 256];
    }
    if (t + 2 < CHUNK_TILES) {
      const char* nxt = reinterpret_cast<const char*>(keys + (size_t)(key0 + 32) * Dd);
      __builtin_prefetch(nxt + tid * 32, 0, 2);       // global_prefetch_b8
    }
    __syncthreads();                                  // tile visible to all waves

    v8f acc0 = {0.f, 0.f, 0.f, 0.f, 0.f, 0.f, 0.f, 0.f};
    v8f acc1 = {0.f, 0.f, 0.f, 0.f, 0.f, 0.f, 0.f, 0.f};
    v8f acc2 = {0.f, 0.f, 0.f, 0.f, 0.f, 0.f, 0.f, 0.f};
    v8f acc3 = {0.f, 0.f, 0.f, 0.f, 0.f, 0.f, 0.f, 0.f};
#pragma unroll
    for (int kk = 0; kk < 8; kk += 4) {
      // A 16x32 bf16 layout: lane<16 row M=lane K 0-7 & 16-23; lane>=16 K 8-15 & 24-31.
      const unsigned short* r0 = sKey + lhalf * Dd + kk * 32 + (hiL ? 8 : 0);
      BFrag a0, a1, a2, a3;
      a0.q[0] = *reinterpret_cast<const U128*>(r0);
      a0.q[1] = *reinterpret_cast<const U128*>(r0 + 16);
      a1.q[0] = *reinterpret_cast<const U128*>(r0 + 32);
      a1.q[1] = *reinterpret_cast<const U128*>(r0 + 48);
      a2.q[0] = *reinterpret_cast<const U128*>(r0 + 64);
      a2.q[1] = *reinterpret_cast<const U128*>(r0 + 80);
      a3.q[0] = *reinterpret_cast<const U128*>(r0 + 96);
      a3.q[1] = *reinterpret_cast<const U128*>(r0 + 112);
      acc0 = __builtin_amdgcn_wmma_f32_16x16x32_bf16(false, a0.v, false, bfrag[kk].v,
                                                     (short)0, acc0, false, false);
      acc1 = __builtin_amdgcn_wmma_f32_16x16x32_bf16(false, a1.v, false, bfrag[kk + 1].v,
                                                     (short)0, acc1, false, false);
      acc2 = __builtin_amdgcn_wmma_f32_16x16x32_bf16(false, a2.v, false, bfrag[kk + 2].v,
                                                     (short)0, acc2, false, false);
      acc3 = __builtin_amdgcn_wmma_f32_16x16x32_bf16(false, a3.v, false, bfrag[kk + 3].v,
                                                     (short)0, acc3, false, false);
    }

    float dv[8];
#pragma unroll
    for (int r = 0; r < 8; ++r)
      dv[r] = (acc0[r] + acc1[r]) + (acc2[r] + acc3[r]);

    // Per-lane pre-filter: if no candidate beats current 6th-best, skip insert
    // (uniform skip when no lane passes; exec-masked correctness otherwise).
    float m = fmaxf(fmaxf(fmaxf(dv[0], dv[1]), fmaxf(dv[2], dv[3])),
                    fmaxf(fmaxf(dv[4], dv[5]), fmaxf(dv[6], dv[7])));
    if (m > tv[TOPP - 1]) {
      const int kb = key0 + (hiL ? 8 : 0);
#pragma unroll
      for (int r = 0; r < 8; ++r) topk_insert(dv[r], kb + r, tv, ti);
    }
  }

  // Merge lane L with lane L+16 (same query, other 8 keys of each tile).
#pragma unroll
  for (int j = 0; j < TOPP; ++j) { sMv[tid * TOPP + j] = tv[j]; sMi[tid * TOPP + j] = ti[j]; }
  __syncthreads();
  if (!hiL) {
    const float* ov = &sMv[(tid + 16) * TOPP];
    const int*   oi = &sMi[(tid + 16) * TOPP];
#pragma unroll
    for (int j = 0; j < TOPP; ++j) topk_insert(ov[j], oi[j], tv, ti);
    const size_t base = ((size_t)qg * NCHUNKS + chunk) * TOPP;
#pragma unroll
    for (int j = 0; j < TOPP; ++j) { pvals[base + j] = tv[j]; pidx[base + j] = ti[j]; }
  }
}

// Merge per-chunk partial top-6 lists; drop rank-0 (self match), emit 5 indices.
__global__ void nrc_merge_topk(const float* __restrict__ pvals, const int* __restrict__ pidx,
                               int* __restrict__ outIdx, int Q) {
  int q = blockIdx.x * blockDim.x + threadIdx.x;
  if (q >= Q) return;
  float tv[TOPP]; int ti[TOPP];
#pragma unroll
  for (int j = 0; j < TOPP; ++j) { tv[j] = -1e30f; ti[j] = 0x7fffffff; }
  for (int c = 0; c < NCHUNKS; ++c) {
    size_t base = ((size_t)q * NCHUNKS + c) * TOPP;
#pragma unroll
    for (int j = 0; j < TOPP; ++j) topk_insert(pvals[base + j], pidx[base + j], tv, ti);
  }
#pragma unroll
  for (int j = 0; j < KK; ++j) outIdx[q * KK + j] = ti[j + 1];
}

// Gather fea_near rows (bf16) to serve as pass-2 queries.
__global__ void nrc_gather_rows(const unsigned short* __restrict__ fb,
                                const int* __restrict__ idx,
                                unsigned short* __restrict__ outq) {
  int r = blockIdx.x, t = threadIdx.x;  // 64 threads * 8B = 512B row
  const U64* s = reinterpret_cast<const U64*>(fb + (size_t)idx[r] * Dd);
  U64* d = reinterpret_cast<U64*>(outq + (size_t)r * Dd);
  d[t] = s[t];
}

// ------------------------------ final loss ---------------------------------

__global__ __launch_bounds__(256) void nrc_loss(const float* __restrict__ p,
                                                const float* __restrict__ sb,
                                                const int* __restrict__ idx_near,
                                                const int* __restrict__ idx_nn,
                                                const int* __restrict__ trg,
                                                float* __restrict__ out) {
  __shared__ float red[256];
  int b = threadIdx.x;
  float pb[Cc];
#pragma unroll
  for (int c = 0; c < Cc; ++c) pb[c] = p[b * Cc + c];

  float acc = 0.f;
  // kl_kk * 0.1 over 25 second-order neighbors
  for (int j = 0; j < KK * KK; ++j) {
    int idx = idx_nn[b * KK * KK + j];
    float kl = 0.f;
#pragma unroll
    for (int c = 0; c < Cc; ++c) { float s = sb[(size_t)idx * Cc + c]; kl += s * (logf(s) - pb[c]); }
    acc += 0.1f * kl;
  }
  // kl_k * weight over 5 first-order neighbors
  for (int k = 0; k < KK; ++k) {
    int idx = idx_near[b * KK + k];
    int match = 0;
#pragma unroll
    for (int j = 0; j < KK; ++j) match += (idx_nn[b * KK * KK + k * KK + j] == trg[b]) ? 1 : 0;
    float w = match > 0 ? (float)match : 0.1f;
    float kl = 0.f;
#pragma unroll
    for (int c = 0; c < Cc; ++c) { float s = sb[(size_t)idx * Cc + c]; kl += s * (logf(s) - pb[c]); }
    acc += w * kl;
  }

  red[b] = acc;
  __syncthreads();
  for (int s = 128; s > 0; s >>= 1) { if (b < s) red[b] += red[b + s]; __syncthreads(); }
  if (b == 0) {
    float loss = red[0] / (float)Bq;
    for (int c = 0; c < Cc; ++c) {
      float m = 0.f;
      for (int bb = 0; bb < Bq; ++bb) m += p[bb * Cc + c];
      m /= (float)Bq;
      loss += m * logf(m + 1e-5f);
    }
    out[0] = loss;
  }
}

// ------------------------------- launcher ----------------------------------

extern "C" void kernel_launch(void* const* d_in, const int* in_sizes, int n_in,
                              void* d_out, int out_size, void* d_ws, size_t ws_size,
                              hipStream_t stream) {
  const float* features    = (const float*)d_in[0];
  const float* predictions = (const float*)d_in[1];
  const float* fea_bank    = (const float*)d_in[2];
  const float* score_bank  = (const float*)d_in[3];
  const int*   trg_idx     = (const int*)d_in[4];
  float* out = (float*)d_out;

  char* ws = (char*)d_ws;
  size_t off = 0;
  auto alloc = [&](size_t bytes) -> void* {
    off = (off + 255) & ~(size_t)255;
    void* ptr = ws + off;
    off += bytes;
    return ptr;
  };

  float*          p_soft   = (float*)alloc((size_t)Bq * Cc * 4);
  unsigned short* fnorm_bf = (unsigned short*)alloc((size_t)Bq * Dd * 2);
  unsigned short* fb_bf    = (unsigned short*)alloc((size_t)Nn * Dd * 2);   // 51.2 MB
  float*          sbuf     = (float*)alloc((size_t)Nn * Cc * 4);            // 2.4 MB
  unsigned short* q2       = (unsigned short*)alloc((size_t)Bq * KK * Dd * 2);
  float*          pvals    = (float*)alloc((size_t)Bq * KK * NCHUNKS * TOPP * 4);
  int*            pidx     = (int*)alloc((size_t)Bq * KK * NCHUNKS * TOPP * 4);
  int*            idx_near = (int*)alloc((size_t)Bq * KK * 4);
  int*            idx_nn   = (int*)alloc((size_t)Bq * KK * KK * 4);
  (void)ws_size; (void)in_sizes; (void)n_in; (void)out_size;

  // 1) softmax(predictions), row-normalize(features) -> bf16
  nrc_softmax<<<1, 256, 0, stream>>>(predictions, p_soft);
  nrc_fnorm<<<Bq, 256, 0, stream>>>(features, fnorm_bf);

  // 2) bank -> bf16 copy; score bank -> f32 copy; deterministic scatter of B rows
  nrc_cvt_bank<<<4096, 256, 0, stream>>>(fea_bank, fb_bf, (Nn * Dd) / 4);
  nrc_copy_f32<<<1024, 256, 0, stream>>>(score_bank, sbuf, Nn * Cc);
  nrc_scatter<<<Bq, 256, 0, stream>>>(trg_idx, fnorm_bf, p_soft, fb_bf, sbuf);

  // 3) pass 1: fnorm @ fb^T with fused top-6 -> idx_near
  nrc_gemm_topk<<<dim3(Bq / QPB, NCHUNKS), 256, 0, stream>>>(fb_bf, fnorm_bf, pvals, pidx);
  nrc_merge_topk<<<1, 256, 0, stream>>>(pvals, pidx, idx_near, Bq);

  // 4) gather fea_near rows as pass-2 queries
  nrc_gather_rows<<<Bq * KK, 64, 0, stream>>>(fb_bf, idx_near, q2);

  // 5) pass 2: fea_near @ fb^T with fused top-6 -> idx_nn
  nrc_gemm_topk<<<dim3((Bq * KK) / QPB, NCHUNKS), 256, 0, stream>>>(fb_bf, q2, pvals, pidx);
  nrc_merge_topk<<<(Bq * KK + 255) / 256, 256, 0, stream>>>(pvals, pidx, idx_nn, Bq * KK);

  // 6) KL losses + marginal entropy -> scalar
  nrc_loss<<<1, 256, 0, stream>>>(p_soft, sbuf, idx_near, idx_nn, trg_idx, out);
}
